// CoarseMatching_15015205666963
// MI455X (gfx1250) — compile-verified, matrix operand order checked by reading, not run on hardware
//
#include <hip/hip_runtime.h>
#include <hip/hip_bf16.h>
#include <math.h>

// Problem constants (from reference)
#define N_BATCH 2
#define L_DIM   6400
#define S_DIM   6400
#define C_DIM   256
#define W_GRID  80
#define BORDER  2
#define THRESH  0.2f
#define RSPLIT  8
#define ROWS_PER_SPLIT (L_DIM / RSPLIT)   // 800

typedef __attribute__((ext_vector_type(16))) __bf16 v16bf;
typedef __attribute__((ext_vector_type(8)))  __bf16 v8bf;
typedef __attribute__((ext_vector_type(8)))  float  v8f;

__device__ __forceinline__ bool border_ok(int idx) {
  int y = idx / W_GRID;
  int x = idx - y * W_GRID;
  return (y >= BORDER) && (y < W_GRID - BORDER) && (x >= BORDER) && (x < W_GRID - BORDER);
}

// ---------------------------------------------------------------- pass 1: fp32 -> bf16
__global__ void cvt_bf16_kernel(const float* __restrict__ x0, const float* __restrict__ x1,
                                __bf16* __restrict__ a, __bf16* __restrict__ b, int n) {
  int i = blockIdx.x * blockDim.x + threadIdx.x;
  if (i < n) {
    a[i] = (__bf16)x0[i];
    b[i] = (__bf16)x1[i];
  }
}

// ---------------------------------------------------------------- pass 2: sim = (A . B^T) / 25.6
// 256 threads = 8 waves; wave grid 2x4 -> workgroup tile 64(L) x 256(S);
// each wave: 2x4 WMMA tiles of 16x16 (32x64 macro-tile) -> 8 WMMAs per 12 b128
// operand loads per K-step (1.5x better operand reuse than 2x2).
__launch_bounds__(256)
__global__ void sim_gemm_kernel(const __bf16* __restrict__ A, const __bf16* __restrict__ B,
                                float* __restrict__ sim) {
  const int lane = threadIdx.x & 31;
  const int wave = threadIdx.x >> 5;
  const int half = lane >> 4;     // lane-half selects K sub-range per ISA layout
  const int r    = lane & 15;     // row within 16x16 tile
  const int l0 = blockIdx.y * 64  + (wave >> 2) * 32;   // 2 wave-rows
  const int s0 = blockIdx.x * 256 + (wave & 3) * 64;    // 4 wave-cols
  const int batch = blockIdx.z;

  const __bf16* a0p = A + ((size_t)batch * L_DIM + (l0 + r)) * C_DIM;
  const __bf16* a1p = a0p + (size_t)16 * C_DIM;
  const __bf16* b0p = B + ((size_t)batch * S_DIM + (s0 + r)) * C_DIM;
  const __bf16* b1p = b0p + (size_t)16 * C_DIM;
  const __bf16* b2p = b0p + (size_t)32 * C_DIM;
  const __bf16* b3p = b0p + (size_t)48 * C_DIM;

  v8f acc00 = {}, acc01 = {}, acc02 = {}, acc03 = {};
  v8f acc10 = {}, acc11 = {}, acc12 = {}, acc13 = {};

  #pragma unroll
  for (int k = 0; k < C_DIM; k += 32) {
    // 16-bit operand layout (ISA 7.12.2): lanes 0-15 hold K = k..k+7 (VGPR0-3)
    // and K = k+16..k+23 (VGPR4-7); lanes 16-31 hold K+8 of each range.
    const int o0 = k + half * 8;
    const int o1 = k + 16 + half * 8;
    union { v16bf v; v8bf h[2]; } a0, a1, b0, b1, b2, b3;
    a0.h[0] = *(const v8bf*)(a0p + o0);  a0.h[1] = *(const v8bf*)(a0p + o1);
    a1.h[0] = *(const v8bf*)(a1p + o0);  a1.h[1] = *(const v8bf*)(a1p + o1);
    b0.h[0] = *(const v8bf*)(b0p + o0);  b0.h[1] = *(const v8bf*)(b0p + o1);
    b1.h[0] = *(const v8bf*)(b1p + o0);  b1.h[1] = *(const v8bf*)(b1p + o1);
    b2.h[0] = *(const v8bf*)(b2p + o0);  b2.h[1] = *(const v8bf*)(b2p + o1);
    b3.h[0] = *(const v8bf*)(b3p + o0);  b3.h[1] = *(const v8bf*)(b3p + o1);

    acc00 = __builtin_amdgcn_wmma_f32_16x16x32_bf16(false, a0.v, false, b0.v, (short)0, acc00, false, false);
    acc01 = __builtin_amdgcn_wmma_f32_16x16x32_bf16(false, a0.v, false, b1.v, (short)0, acc01, false, false);
    acc02 = __builtin_amdgcn_wmma_f32_16x16x32_bf16(false, a0.v, false, b2.v, (short)0, acc02, false, false);
    acc03 = __builtin_amdgcn_wmma_f32_16x16x32_bf16(false, a0.v, false, b3.v, (short)0, acc03, false, false);
    acc10 = __builtin_amdgcn_wmma_f32_16x16x32_bf16(false, a1.v, false, b0.v, (short)0, acc10, false, false);
    acc11 = __builtin_amdgcn_wmma_f32_16x16x32_bf16(false, a1.v, false, b1.v, (short)0, acc11, false, false);
    acc12 = __builtin_amdgcn_wmma_f32_16x16x32_bf16(false, a1.v, false, b2.v, (short)0, acc12, false, false);
    acc13 = __builtin_amdgcn_wmma_f32_16x16x32_bf16(false, a1.v, false, b3.v, (short)0, acc13, false, false);
  }

  // C/D layout: VGPR e, lane -> row = e + 8*half, col = lane&15
  const float scl = 1.0f / (256.0f * 0.1f);
  float* out = sim + ((size_t)batch * L_DIM + l0 + 8 * half) * S_DIM + s0 + r;
  #pragma unroll
  for (int e = 0; e < 8; ++e) {
    float* p0 = out + (size_t)e * S_DIM;                     // row tile 0
    float* p1 = p0 + (size_t)16 * S_DIM;                     // row tile 1
    p0[0]  = acc00[e] * scl;
    p0[16] = acc01[e] * scl;
    p0[32] = acc02[e] * scl;
    p0[48] = acc03[e] * scl;
    p1[0]  = acc10[e] * scl;
    p1[16] = acc11[e] * scl;
    p1[32] = acc12[e] * scl;
    p1[48] = acc13[e] * scl;
  }
}

// ---------------------------------------------------------------- pass 3: row softmax stats
__global__ void row_stats_kernel(const float* __restrict__ sim,
                                 float* __restrict__ rmax, float* __restrict__ rinv) {
  const int row = blockIdx.x, batch = blockIdx.y;
  const float* p = sim + ((size_t)batch * L_DIM + row) * S_DIM;
  float m = -INFINITY, d = 0.f;
  for (int s = threadIdx.x; s < S_DIM; s += 256) {
    float v = p[s];
    if (v > m) { d = d * __expf(m - v) + 1.f; m = v; }
    else       { d += __expf(v - m); }
  }
  __shared__ float sm[256], sd[256];
  sm[threadIdx.x] = m; sd[threadIdx.x] = d;
  __syncthreads();
  for (int off = 128; off > 0; off >>= 1) {
    if (threadIdx.x < off) {
      float m1 = sm[threadIdx.x], d1 = sd[threadIdx.x];
      float m2 = sm[threadIdx.x + off], d2 = sd[threadIdx.x + off];
      float M = fmaxf(m1, m2);
      sm[threadIdx.x] = M;
      sd[threadIdx.x] = d1 * __expf(m1 - M) + d2 * __expf(m2 - M);
    }
    __syncthreads();
  }
  if (threadIdx.x == 0) {
    rmax[batch * L_DIM + row] = sm[0];
    rinv[batch * L_DIM + row] = 1.0f / sd[0];
  }
}

// ---------------------------------------------------------------- pass 4a: column stats partials
__global__ void col_partial_kernel(const float* __restrict__ sim,
                                   float* __restrict__ pm, float* __restrict__ pd) {
  const int s = blockIdx.x * 256 + threadIdx.x;
  const int chunk = blockIdx.y, batch = blockIdx.z;
  const float* p = sim + (size_t)batch * L_DIM * S_DIM + (size_t)chunk * ROWS_PER_SPLIT * S_DIM + s;
  float m = -INFINITY, d = 0.f;
  for (int l = 0; l < ROWS_PER_SPLIT; ++l) {
    float v = p[(size_t)l * S_DIM];
    if (v > m) { d = d * __expf(m - v) + 1.f; m = v; }
    else       { d += __expf(v - m); }
  }
  size_t o = ((size_t)chunk * N_BATCH + batch) * S_DIM + s;
  pm[o] = m; pd[o] = d;
}

// ---------------------------------------------------------------- pass 4b: column stats combine
__global__ void col_combine_kernel(const float* __restrict__ pm, const float* __restrict__ pd,
                                   float* __restrict__ cmax, float* __restrict__ cinv) {
  const int s = blockIdx.x * 256 + threadIdx.x;
  const int batch = blockIdx.y;
  float m = -INFINITY, d = 0.f;
  for (int c = 0; c < RSPLIT; ++c) {
    size_t o = ((size_t)c * N_BATCH + batch) * S_DIM + s;
    float m2 = pm[o], d2 = pd[o];
    float M = fmaxf(m, m2);
    d = d * __expf(m - M) + d2 * __expf(m2 - M);
    m = M;
  }
  cmax[batch * S_DIM + s] = m;
  cinv[batch * S_DIM + s] = 1.0f / d;
}

// ---------------------------------------------------------------- pass 5: conf (in place) + row max of conf
__global__ void conf_rowmax_kernel(float* __restrict__ sim,
                                   const float* __restrict__ rmax, const float* __restrict__ rinv,
                                   const float* __restrict__ cmax, const float* __restrict__ cinv,
                                   float* __restrict__ rcmax) {
  const int row = blockIdx.x, batch = blockIdx.y;
  const float rm = rmax[batch * L_DIM + row];
  const float ri = rinv[batch * L_DIM + row];
  float* p = sim + ((size_t)batch * L_DIM + row) * S_DIM;
  const float* cm = cmax + batch * S_DIM;
  const float* ci = cinv + batch * S_DIM;
  float tmax = 0.f;
  for (int s = threadIdx.x; s < S_DIM; s += 256) {
    float v = p[s];
    // exp(v-rm)*ri * exp(v-cm)*ci, fused: exponent <= 0 -> no overflow
    float c = __expf(v + v - rm - cm[s]) * (ri * ci[s]);
    p[s] = c;
    tmax = fmaxf(tmax, c);
  }
  __shared__ float sm[256];
  sm[threadIdx.x] = tmax; __syncthreads();
  for (int off = 128; off > 0; off >>= 1) {
    if (threadIdx.x < off) sm[threadIdx.x] = fmaxf(sm[threadIdx.x], sm[threadIdx.x + off]);
    __syncthreads();
  }
  if (threadIdx.x == 0) rcmax[batch * L_DIM + row] = sm[0];
}

// ---------------------------------------------------------------- pass 6: column max of conf
__global__ void colmax_partial_kernel(const float* __restrict__ conf, float* __restrict__ pcm) {
  const int s = blockIdx.x * 256 + threadIdx.x;
  const int chunk = blockIdx.y, batch = blockIdx.z;
  const float* p = conf + (size_t)batch * L_DIM * S_DIM + (size_t)chunk * ROWS_PER_SPLIT * S_DIM + s;
  float m = 0.f;
  for (int l = 0; l < ROWS_PER_SPLIT; ++l) m = fmaxf(m, p[(size_t)l * S_DIM]);
  pcm[((size_t)chunk * N_BATCH + batch) * S_DIM + s] = m;
}

__global__ void colmax_combine_kernel(const float* __restrict__ pcm, float* __restrict__ ccmax) {
  const int s = blockIdx.x * 256 + threadIdx.x;
  const int batch = blockIdx.y;
  float m = 0.f;
  for (int c = 0; c < RSPLIT; ++c)
    m = fmaxf(m, pcm[((size_t)c * N_BATCH + batch) * S_DIM + s]);
  ccmax[batch * S_DIM + s] = m;
}

// ---------------------------------------------------------------- pass 7: mask + scores
__global__ void final_kernel(const float* __restrict__ conf,
                             const float* __restrict__ rcmax, const float* __restrict__ ccmax,
                             float* __restrict__ mask, float* __restrict__ scores) {
  const int row = blockIdx.x, batch = blockIdx.y;
  const bool rok = border_ok(row);
  const float rm = rcmax[batch * L_DIM + row];
  const size_t base = ((size_t)batch * L_DIM + row) * S_DIM;
  const float* cc = ccmax + batch * S_DIM;
  for (int s = threadIdx.x; s < S_DIM; s += 256) {
    float c = conf[base + s];
    bool ok = (c > THRESH) && rok && border_ok(s) && (c == rm) && (c == cc[s]);
    mask[base + s]   = ok ? 1.0f : 0.0f;
    scores[base + s] = ok ? c : 0.0f;
  }
}

// ----------------------------------------------------------------
extern "C" void kernel_launch(void* const* d_in, const int* in_sizes, int n_in,
                              void* d_out, int out_size, void* d_ws, size_t ws_size,
                              hipStream_t stream) {
  (void)in_sizes; (void)n_in; (void)out_size; (void)ws_size;
  const float* x0 = (const float*)d_in[0];
  const float* x1 = (const float*)d_in[1];

  float* out = (float*)d_out;
  const size_t conf_elems = (size_t)N_BATCH * L_DIM * S_DIM;
  float* conf   = out;                    // pass2 writes sim here, pass5 rewrites as conf
  float* mask   = out + conf_elems;       // final output; used as bf16 staging until pass 7
  float* scores = out + 2 * conf_elems;

  // bf16 operands staged inside the mask region (13 MB << 327 MB), consumed by GEMM only
  __bf16* Abf = (__bf16*)mask;
  __bf16* Bbf = Abf + (size_t)N_BATCH * L_DIM * C_DIM;

  // small stats arrays in workspace (~1.5 MB)
  float* ws = (float*)d_ws;
  const int NL = N_BATCH * L_DIM, NS = N_BATCH * S_DIM;
  float* rmax  = ws;
  float* rinv  = rmax  + NL;
  float* cmax  = rinv  + NL;
  float* cinv  = cmax  + NS;
  float* rcmax = cinv  + NS;
  float* ccmax = rcmax + NL;
  float* pm    = ccmax + NS;
  float* pd    = pm + (size_t)RSPLIT * NS;
  float* pcm   = pd + (size_t)RSPLIT * NS;

  const int nconv = N_BATCH * L_DIM * C_DIM;
  cvt_bf16_kernel<<<(nconv + 255) / 256, 256, 0, stream>>>(x0, x1, Abf, Bbf, nconv);

  sim_gemm_kernel<<<dim3(S_DIM / 256, L_DIM / 64, N_BATCH), 256, 0, stream>>>(Abf, Bbf, conf);

  row_stats_kernel<<<dim3(L_DIM, N_BATCH), 256, 0, stream>>>(conf, rmax, rinv);
  col_partial_kernel<<<dim3(S_DIM / 256, RSPLIT, N_BATCH), 256, 0, stream>>>(conf, pm, pd);
  col_combine_kernel<<<dim3(S_DIM / 256, N_BATCH), 256, 0, stream>>>(pm, pd, cmax, cinv);

  conf_rowmax_kernel<<<dim3(L_DIM, N_BATCH), 256, 0, stream>>>(conf, rmax, rinv, cmax, cinv, rcmax);
  colmax_partial_kernel<<<dim3(S_DIM / 256, RSPLIT, N_BATCH), 256, 0, stream>>>(conf, pcm);
  colmax_combine_kernel<<<dim3(S_DIM / 256, N_BATCH), 256, 0, stream>>>(pcm, ccmax);

  final_kernel<<<dim3(L_DIM, N_BATCH), 256, 0, stream>>>(conf, rcmax, ccmax, mask, scores);
}
